// CrossCosineEmbeddingLoss_36610301231301
// MI455X (gfx1250) — matrix operand order repaired and checked by reading, not compiled, not run on hardware
//
#include <hip/hip_runtime.h>

// Problem constants (reference: x,y are [8192,1024] f32; loss is scalar f32)
#define NROWS 8192
#define KDIM  1024

// GEMM tiling
#define BM 128
#define BN 128
#define BK 32
#define LDA 40    // f16 elements per sA row (32 + 8 pad) -> 80B stride
#define LDB 136   // f16 elements per sB row (128 + 8 pad) -> 272B stride

typedef __attribute__((ext_vector_type(16))) _Float16 v16h;
typedef __attribute__((ext_vector_type(4)))  _Float16 v4h;
typedef __attribute__((ext_vector_type(8)))  float    v8f;

__device__ __forceinline__ float wave_reduce_add(float v) {
#pragma unroll
  for (int off = 16; off > 0; off >>= 1) v += __shfl_down(v, off, 32);
  return v;
}

// Valid result in thread 0. Block must be 256 threads (8 wave32).
__device__ __forceinline__ float block_reduce_add_256(float v) {
  __shared__ float red[8];
  v = wave_reduce_add(v);
  const int wave = threadIdx.x >> 5;
  const int lane = threadIdx.x & 31;
  if (lane == 0) red[wave] = v;
  __syncthreads();
  float t = (threadIdx.x < 8) ? red[threadIdx.x] : 0.0f;
  if (wave == 0) {
#pragma unroll
    for (int off = 4; off > 0; off >>= 1) t += __shfl_down(t, off, 32);
  }
  return t;
}

// --- CDNA5 async global->LDS copy (ASYNCcnt-tracked, no VGPR staging) -----
// VGLOBAL async encoding: VDST = LDS byte address (32-bit VGPR),
// VADDR = 64-bit global address, SADDR = off. Flat->LDS truncation gives the
// workgroup-relative LDS offset from a generic pointer.
__device__ __forceinline__ void async_ld_b128(unsigned lds_addr,
                                              const _Float16* g) {
  asm volatile("global_load_async_to_lds_b128 %0, %1, off"
               :
               : "v"(lds_addr), "v"((unsigned long long)(uintptr_t)g)
               : "memory");
}
__device__ __forceinline__ void wait_async0() {
  asm volatile("s_wait_asynccnt 0x0" ::: "memory");
}
__device__ __forceinline__ unsigned lds_addr_of(const void* p) {
  return (unsigned)(uintptr_t)p;
}

// --- Kernel 1: normalize x rows, emit f16 row-major (WMMA A operand) ------
__global__ __launch_bounds__(256) void rownorm_x_kernel(
    const float* __restrict__ x, _Float16* __restrict__ xn) {
  const int row = blockIdx.x;
  const float4* xr = (const float4*)(x + (size_t)row * KDIM);
  float4 v = xr[threadIdx.x];                 // 256 threads * 4 = 1024
  float s = v.x * v.x + v.y * v.y + v.z * v.z + v.w * v.w;
  float tot = block_reduce_add_256(s);
  __shared__ float scale_s;
  if (threadIdx.x == 0) scale_s = 1.0f / fmaxf(sqrtf(tot), 1e-8f);
  __syncthreads();
  const float sc = scale_s;
  v4h o;
  o[0] = (_Float16)(v.x * sc);
  o[1] = (_Float16)(v.y * sc);
  o[2] = (_Float16)(v.z * sc);
  o[3] = (_Float16)(v.w * sc);
  *(v4h*)(xn + (size_t)row * KDIM + threadIdx.x * 4) = o;
}

// --- Kernel 2: per-row inverse norm of y ----------------------------------
__global__ __launch_bounds__(256) void rownorm_y_kernel(
    const float* __restrict__ y, float* __restrict__ yscale) {
  const int row = blockIdx.x;
  const float4* yr = (const float4*)(y + (size_t)row * KDIM);
  float4 v = yr[threadIdx.x];
  float s = v.x * v.x + v.y * v.y + v.z * v.z + v.w * v.w;
  float tot = block_reduce_add_256(s);
  if (threadIdx.x == 0) yscale[row] = 1.0f / fmaxf(sqrtf(tot), 1e-8f);
}

// --- Kernel 3: tiled transpose+scale: ynT[k][j] = y[j][k] / ||y_j|| (f16) -
__global__ __launch_bounds__(256) void transpose_scale_y_kernel(
    const float* __restrict__ y, const float* __restrict__ yscale,
    _Float16* __restrict__ ynT) {
  __shared__ float tile[32][33];
  __shared__ float sc[32];
  const int j0 = blockIdx.x * 32;   // row block of y
  const int k0 = blockIdx.y * 32;   // col block of y
  const int tx = threadIdx.x & 31;
  const int ty = threadIdx.x >> 5;  // 0..7
  if (threadIdx.x < 32) sc[threadIdx.x] = yscale[j0 + threadIdx.x];
#pragma unroll
  for (int r = 0; r < 32; r += 8)
    tile[ty + r][tx] = y[(size_t)(j0 + ty + r) * KDIM + k0 + tx];
  __syncthreads();
#pragma unroll
  for (int r = 0; r < 32; r += 8)
    ynT[(size_t)(k0 + ty + r) * NROWS + j0 + tx] =
        (_Float16)(tile[tx][ty + r] * sc[tx]);
}

// --- Kernel 4: WMMA GEMM (128x128 tile), double-buffered LDS fed by async
// global->LDS copies, fused hinge-loss epilogue.
// A: xn  row-major [NROWS][KDIM] f16 ; Bt: ynT K-major [KDIM][NROWS] f16.
// 8 waves: 2 (M) x 4 (N); wave tile 64x32 = 4x2 WMMA accumulators.
__global__ __launch_bounds__(256) void gemm_loss_kernel(
    const _Float16* __restrict__ A, const _Float16* __restrict__ Bt,
    float* __restrict__ partial) {
  __shared__ __align__(16) _Float16 sA[2][BM * LDA];
  __shared__ __align__(16) _Float16 sB[2][BK * LDB];

  const int tid  = threadIdx.x;
  const int wave = tid >> 5;
  const int lane = tid & 31;
  const int waveM = wave >> 2;  // 0..1 -> 64 rows
  const int waveN = wave & 3;   // 0..3 -> 32 cols
  const int rowBase = blockIdx.y * BM;
  const int colBase = blockIdx.x * BN;

  v8f acc[4][2] = {};

  // Staging assignment: 256 threads move 8KB A + 8KB B per K-step,
  // 2 x b128 per thread per operand, direct to LDS via ASYNC ops.
  const int aRow  = tid & 127;  // A tile row
  const int aHalf = tid >> 7;   // which 16-element chunk of the 32-wide k
  const int bK    = tid & 31;   // B tile k-row
  const int bQ    = tid >> 5;   // which 16-element chunk of 128 cols

  const _Float16* aG = A + (size_t)(rowBase + aRow) * KDIM + aHalf * 16;
  const _Float16* bG = Bt + (size_t)bK * NROWS + colBase + bQ * 16;
  const unsigned dA0 = lds_addr_of(&sA[0][aRow * LDA + aHalf * 16]);
  const unsigned dA1 = lds_addr_of(&sA[1][aRow * LDA + aHalf * 16]);
  const unsigned dB0 = lds_addr_of(&sB[0][bK * LDB + bQ * 16]);
  const unsigned dB1 = lds_addr_of(&sB[1][bK * LDB + bQ * 16]);

  // A fragment addressing per the 16-bit A-matrix 16x32 VGPR layout:
  // lanes 0-15: M=lane, K = 0..7 then 16..23 ; lanes 16-31: M=lane-16,
  // K = 8..15 then 24..31.
  const int mrow  = lane & 15;
  const int khalf = lane >> 4;

  // Prologue: tile 0 -> buffer 0.
  async_ld_b128(dA0, aG);
  async_ld_b128(dA0 + 16, aG + 8);
  async_ld_b128(dB0, bG);
  async_ld_b128(dB0 + 16, bG + 8);

  for (int kb = 0; kb < KDIM; kb += BK) {
    const int cur = (kb / BK) & 1;
    // Tile kb (issued last iteration) resident for this wave...
    wait_async0();
    // ...and for every wave in the workgroup.
    __syncthreads();
    // Prefetch tile kb+1 into the other buffer. Safe: every wave's reads of
    // that buffer (iteration kb-1) completed before this barrier.
    if (kb + BK < KDIM) {
      const _Float16* aN = aG + (kb + BK);
      const _Float16* bN = bG + (size_t)(kb + BK) * NROWS;
      const unsigned nA = cur ? dA0 : dA1;
      const unsigned nB = cur ? dB0 : dB1;
      async_ld_b128(nA, aN);
      async_ld_b128(nA + 16, aN + 8);
      async_ld_b128(nB, bN);
      async_ld_b128(nB + 16, bN + 8);
    }

    const _Float16* sAc = sA[cur];
    const _Float16* sBc = sB[cur];
    union Frag { uint4 u[2]; v16h h; };
    Frag af[4], bf[2];
#pragma unroll
    for (int ni = 0; ni < 2; ++ni) {
      // B layout: N = lane&15; K = khalf*8 .. +7 then 16+khalf*8 .. +7.
      const _Float16* q = &sBc[waveN * 32 + ni * 16 + mrow];
#pragma unroll
      for (int s = 0; s < 8; ++s) {
        bf[ni].h[s]     = q[(khalf * 8 + s) * LDB];
        bf[ni].h[s + 8] = q[(16 + khalf * 8 + s) * LDB];
      }
    }
#pragma unroll
    for (int mi = 0; mi < 4; ++mi) {
      const _Float16* p = &sAc[(waveM * 64 + mi * 16 + mrow) * LDA + khalf * 8];
      af[mi].u[0] = *(const uint4*)(p);        // K = khalf*8 .. +7
      af[mi].u[1] = *(const uint4*)(p + 16);   // K = 16+khalf*8 .. +7
    }
#pragma unroll
    for (int mi = 0; mi < 4; ++mi)
#pragma unroll
      for (int ni = 0; ni < 2; ++ni)
        acc[mi][ni] = __builtin_amdgcn_wmma_f32_16x16x32_f16(
            false, af[mi].h, false, bf[ni].h, (short)0, acc[mi][ni],
            false, false);
  }

  // Fused loss epilogue. C/D layout: element (M = v + 8*(lane>>4),
  // N = lane&15) in VGPR v.
  float lsum = 0.0f;
  const int nIdx  = lane & 15;
  const int mOff8 = (lane >> 4) * 8;
#pragma unroll
  for (int mi = 0; mi < 4; ++mi) {
    const int iBase = rowBase + waveM * 64 + mi * 16 + mOff8;
#pragma unroll
    for (int ni = 0; ni < 2; ++ni) {
      const int jj = colBase + waveN * 32 + ni * 16 + nIdx;
#pragma unroll
      for (int v = 0; v < 8; ++v) {
        const float s = acc[mi][ni][v];
        lsum += (iBase + v == jj) ? (1.0f - s) : fmaxf(s, 0.0f);
      }
    }
  }
  float tot = block_reduce_add_256(lsum);
  if (tid == 0) partial[blockIdx.y * gridDim.x + blockIdx.x] = tot;
}

// --- Kernel 5: deterministic final reduction ------------------------------
__global__ __launch_bounds__(256) void finalize_kernel(
    const float* __restrict__ partial, float* __restrict__ out) {
  float s = 0.0f;
  for (int i = threadIdx.x; i < (NROWS / BM) * (NROWS / BN); i += 256)
    s += partial[i];
  float tot = block_reduce_add_256(s);
  if (threadIdx.x == 0)
    out[0] = tot * (1.0f / ((float)NROWS * (float)NROWS));
}

extern "C" void kernel_launch(void* const* d_in, const int* in_sizes, int n_in,
                              void* d_out, int out_size, void* d_ws,
                              size_t ws_size, hipStream_t stream) {
  const float* x = (const float*)d_in[0];
  const float* y = (const float*)d_in[1];
  float* out = (float*)d_out;

  // Workspace layout (requires ~33.6 MB):
  //   [0, 16MB)      xn   f16 [NROWS][KDIM]
  //   [16MB, 32MB)   ynT  f16 [KDIM][NROWS]
  //   [32MB, +32KB)  yscale f32 [NROWS]
  //   then           partial f32 [64*64]
  char* ws = (char*)d_ws;
  _Float16* xn   = (_Float16*)(ws);
  _Float16* ynT  = (_Float16*)(ws + (size_t)NROWS * KDIM * 2);
  float*    yscl = (float*)(ws + (size_t)NROWS * KDIM * 4);
  float*    part = yscl + NROWS;

  rownorm_x_kernel<<<NROWS, 256, 0, stream>>>(x, xn);
  rownorm_y_kernel<<<NROWS, 256, 0, stream>>>(y, yscl);
  transpose_scale_y_kernel<<<dim3(NROWS / 32, KDIM / 32), 256, 0, stream>>>(
      y, yscl, ynT);
  gemm_loss_kernel<<<dim3(NROWS / BN, NROWS / BM), 256, 0, stream>>>(xn, ynT,
                                                                     part);
  finalize_kernel<<<1, 256, 0, stream>>>(part, out);
}
